// Head_33277406609450
// MI455X (gfx1250) — matrix operand order repaired and checked by reading, not compiled, not run on hardware
//
#include <hip/hip_runtime.h>
#include <hip/hip_bf16.h>

// ---------------------------------------------------------------------------
// Causal single-head attention, B=4 T=2048 E=1024 HS=64, f32 in/out.
// bf16 WMMA 16x16x32, f32 accumulation, flash softmax.
//  - projection: one wave per (row-tile, matrix); K-loop unrolled x2 with
//    ping-pong A/B operand buffers (no rotation movs, loads overlap 4 WMMAs).
//  - attention: 2 waves per query tile (split key range + LDS flash-merge);
//    K/V tiles ping-pong double-buffered across key blocks.
// ---------------------------------------------------------------------------

#define BB 4
#define TT 2048
#define EE 1024
#define HS 64

typedef __bf16  v16bf __attribute__((ext_vector_type(16)));
typedef float   v8f   __attribute__((ext_vector_type(8)));

__device__ __forceinline__ __bf16 tobf(float f) { return (__bf16)f; }

// 16 contiguous bf16 (32B) as two b128 loads
__device__ __forceinline__ v16bf load16(const __bf16* p) {
    union { struct { uint4 a, b; } u; v16bf v; } cv;
    cv.u.a = *(const uint4*)p;
    cv.u.b = *(const uint4*)(p + 8);
    return cv.v;
}

// A-tile gather: elems 0..7 at p[0..7], elems 8..15 at p[16..23]
__device__ __forceinline__ v16bf load8x2(const __bf16* p) {
    union { struct { uint4 a, b; } u; v16bf v; } cv;
    cv.u.a = *(const uint4*)p;
    cv.u.b = *(const uint4*)(p + 16);
    return cv.v;
}

__device__ __forceinline__ v8f wmma_bf16(const v16bf& a, const v16bf& b, const v8f& c) {
    return __builtin_amdgcn_wmma_f32_16x16x32_bf16(false, a, false, b, (short)0, c,
                                                   false, false);
}

// ---------------------------------------------------------------------------
// Kernel 0: weights -> bf16, TRANSPOSED: Wt[m][n][e] = W[e][n]; Wq scaled by 8.
// ---------------------------------------------------------------------------
__global__ void convw_kernel(const float* __restrict__ Wq,
                             const float* __restrict__ Wk,
                             const float* __restrict__ Wv,
                             __bf16* __restrict__ Wt) {
    int i = blockIdx.x * blockDim.x + threadIdx.x;   // i = e*HS + n
    if (i < EE * HS) {
        int e = i >> 6, n = i & 63;
        Wt[(size_t)(0 * HS + n) * EE + e] = tobf(Wq[i] * 8.0f);
        Wt[(size_t)(1 * HS + n) * EE + e] = tobf(Wk[i]);
        Wt[(size_t)(2 * HS + n) * EE + e] = tobf(Wv[i]);
    }
}

// ---------------------------------------------------------------------------
// Kernel 1: projections. One wave per (row-tile, matrix m): 1536 waves.
// K-loop unrolled x2 with ping-pong operand buffers.
// ---------------------------------------------------------------------------
__global__ __launch_bounds__(128, 1) void proj_kernel(
        const float*  __restrict__ x,
        const __bf16* __restrict__ Wt,     // [3][HS][E] bf16, Wq pre-scaled
        __bf16* __restrict__ qb,
        __bf16* __restrict__ kb,
        __bf16* __restrict__ vt) {
    const int wave = (blockIdx.x * blockDim.x + threadIdx.x) >> 5;   // 0..1535
    const int lane = threadIdx.x & 31;
    const int half = lane >> 4;
    const int lrow = lane & 15;
    const int m       = wave % 3;          // adjacent waves share the x row-tile
    const int rowbase = (wave / 3) * 16;

    const float*  xrow  = x  + (size_t)(rowbase + lrow) * EE + half * 8;
    const __bf16* wbase = Wt + (size_t)(m * HS + lrow) * EE + half * 8;

    auto loadA = [&](int kc, float4* f) {
        const float* xn = xrow + kc;
        f[0] = *(const float4*)(xn);
        f[1] = *(const float4*)(xn + 4);
        f[2] = *(const float4*)(xn + 16);
        f[3] = *(const float4*)(xn + 20);
    };
    auto loadB = [&](int kc, v16bf* bt) {
#pragma unroll
        for (int nt = 0; nt < 4; ++nt)
            bt[nt] = load8x2(wbase + (size_t)(nt * 16) * EE + kc);
    };
    auto cvtA = [&](const float4* f) {
        v16bf a;
        a[0]  = tobf(f[0].x); a[1]  = tobf(f[0].y); a[2]  = tobf(f[0].z); a[3]  = tobf(f[0].w);
        a[4]  = tobf(f[1].x); a[5]  = tobf(f[1].y); a[6]  = tobf(f[1].z); a[7]  = tobf(f[1].w);
        a[8]  = tobf(f[2].x); a[9]  = tobf(f[2].y); a[10] = tobf(f[2].z); a[11] = tobf(f[2].w);
        a[12] = tobf(f[3].x); a[13] = tobf(f[3].y); a[14] = tobf(f[3].z); a[15] = tobf(f[3].w);
        return a;
    };

    v8f acc[4];
#pragma unroll
    for (int t = 0; t < 4; ++t) acc[t] = (v8f)0.0f;

    float4 fA[4], fB[4];
    v16bf  btA[4], btB[4];
    loadA(0, fA);
    loadB(0, btA);

    for (int kc = 0; kc < EE; kc += 64) {
        // ---- phase 0: refill buffer B (kc+32), consume buffer A (kc) --------
        loadA(kc + 32, fB);
        loadB(kc + 32, btB);
        const v16bf a0 = cvtA(fA);
        __builtin_amdgcn_sched_barrier(0);
#pragma unroll
        for (int t = 0; t < 4; ++t) acc[t] = wmma_bf16(a0, btA[t], acc[t]);

        // ---- phase 1: refill buffer A (kc+64, wrap), consume buffer B -------
        const int kn = (kc + 64 < EE) ? kc + 64 : 0;
        loadA(kn, fA);
        loadB(kn, btA);
        const v16bf a1 = cvtA(fB);
        __builtin_amdgcn_sched_barrier(0);
#pragma unroll
        for (int t = 0; t < 4; ++t) acc[t] = wmma_bf16(a1, btB[t], acc[t]);
    }

    // ---- store: m==0 -> q row-major, m==1 -> k row-major, m==2 -> v^T -------
    if (m == 0) {
#pragma unroll
        for (int nt = 0; nt < 4; ++nt)
#pragma unroll
            for (int vv = 0; vv < 8; ++vv)
                qb[(size_t)(rowbase + vv + half * 8) * HS + nt * 16 + lrow] =
                    tobf(acc[nt][vv]);
    } else if (m == 1) {
#pragma unroll
        for (int nt = 0; nt < 4; ++nt)
#pragma unroll
            for (int vv = 0; vv < 8; ++vv)
                kb[(size_t)(rowbase + vv + half * 8) * HS + nt * 16 + lrow] =
                    tobf(acc[nt][vv]);
    } else {
        const int bidx = rowbase >> 11;            // /TT
        const int trow = rowbase & (TT - 1);
#pragma unroll
        for (int nt = 0; nt < 4; ++nt)
#pragma unroll
            for (int vv = 0; vv < 8; ++vv)
                vt[(size_t)bidx * HS * TT + (size_t)(nt * 16 + lrow) * TT
                   + trow + vv + half * 8] = tobf(acc[nt][vv]);
    }
}

// ---------------------------------------------------------------------------
// Kernel 2: causal flash attention. One BLOCK (2 waves) per 16-query tile:
// wave 0 takes key blocks [0, nb/2), wave 1 takes [nb/2, nb) incl. diagonal;
// partial (m, l, O) states merged through LDS, wave 0 writes output.
// K/V tiles ping-pong double-buffered across key blocks.
// ---------------------------------------------------------------------------
__global__ __launch_bounds__(64, 1) void attn_kernel(
        const __bf16* __restrict__ qb,
        const __bf16* __restrict__ kb,
        const __bf16* __restrict__ vt,     // [b][hs][t]
        float* __restrict__ out) {
    const int wid  = threadIdx.x >> 5;                 // 0 or 1
    const int lane = threadIdx.x & 31;
    const int half = lane >> 4;
    const int lrow = lane & 15;
    const int tile = blockIdx.x;                       // 0 .. B*T/16-1
    const int b    = tile >> 7;
    const int qbase = (tile & 127) * 16;
    const size_t base  = (size_t)b * TT * HS;          // q/k row-major base
    const size_t vbase = (size_t)b * HS * TT;          // v transposed base

    __shared__ __align__(16) __bf16 Psh[2][16 * 32];   // per-wave P staging
    __shared__ __align__(16) float  Osh[16][HS];       // wave-1 partial O
    __shared__ float Msh[16], Lsh[16];                 // wave-1 partial m, l
    __bf16* P = &Psh[wid][0];

    // ---- Q tile: two A tiles covering HS=64 ---------------------------------
    const __bf16* qr = qb + base + (size_t)(qbase + lrow) * HS;
    v16bf qa[2];
#pragma unroll
    for (int c = 0; c < 2; ++c) qa[c] = load16(qr + c * 32 + half * 16);

    v8f o[4];
#pragma unroll
    for (int nt = 0; nt < 4; ++nt) o[nt] = (v8f)0.0f;
    float mrow[8], lsum[8];
#pragma unroll
    for (int vv = 0; vv < 8; ++vv) { mrow[vv] = -1e30f; lsum[vv] = 0.0f; }

    auto loadKV = [&](int blk, v16bf* kt_, v16bf* vt_) {
        const int s0 = blk * 32;
#pragma unroll
        for (int kt = 0; kt < 2; ++kt)
#pragma unroll
            for (int c = 0; c < 2; ++c)
                kt_[kt * 2 + c] = load16(kb + base
                                         + (size_t)(s0 + kt * 16 + lrow) * HS
                                         + c * 32 + half * 16);
#pragma unroll
        for (int nt = 0; nt < 4; ++nt)
            vt_[nt] = load16(vt + vbase + (size_t)(nt * 16 + lrow) * TT
                             + s0 + half * 16);
    };

    auto process = [&](int blk, const v16bf* kbt, const v16bf* vbt) {
        const int s0 = blk * 32;
        v8f s[2]; s[0] = (v8f)0.0f; s[1] = (v8f)0.0f;
#pragma unroll
        for (int kt = 0; kt < 2; ++kt)
#pragma unroll
            for (int c = 0; c < 2; ++c)
                s[kt] = wmma_bf16(qa[c], kbt[kt * 2 + c], s[kt]);

        float corr[8];
#pragma unroll
        for (int vv = 0; vv < 8; ++vv) {
            const int qi = qbase + vv + half * 8;
            float x0 = s[0][vv], x1 = s[1][vv];
            if (s0 + lrow > qi)      x0 = -1e30f;
            if (s0 + 16 + lrow > qi) x1 = -1e30f;
            float rmax = fmaxf(x0, x1);
#pragma unroll
            for (int msk = 1; msk < 16; msk <<= 1)
                rmax = fmaxf(rmax, __shfl_xor(rmax, msk, 32));
            const float mnew = fmaxf(mrow[vv], rmax);
            const float cr   = __expf(mrow[vv] - mnew);
            const float p0   = __expf(x0 - mnew);
            const float p1   = __expf(x1 - mnew);
            float ps = p0 + p1;
#pragma unroll
            for (int msk = 1; msk < 16; msk <<= 1)
                ps += __shfl_xor(ps, msk, 32);
            lsum[vv] = lsum[vv] * cr + ps;
            mrow[vv] = mnew;
            corr[vv] = cr;
            P[(vv + half * 8) * 32 + lrow]      = tobf(p0);
            P[(vv + half * 8) * 32 + 16 + lrow] = tobf(p1);
        }
#pragma unroll
        for (int nt = 0; nt < 4; ++nt)
#pragma unroll
            for (int vv = 0; vv < 8; ++vv) o[nt][vv] *= corr[vv];

        __builtin_amdgcn_wave_barrier();
        asm volatile("s_wait_dscnt 0" ::: "memory");

        const v16bf pa = load16(P + lrow * 32 + half * 16);
#pragma unroll
        for (int nt = 0; nt < 4; ++nt)
            o[nt] = wmma_bf16(pa, vbt[nt], o[nt]);

        __builtin_amdgcn_wave_barrier();
    };

    const int nb     = ((qbase + 15) >> 5) + 1;        // #key blocks of 32
    const int bStart = (wid == 0) ? 0 : (nb >> 1);
    const int bEnd   = (wid == 0) ? (nb >> 1) : nb;

    v16bf k0[4], v0[4], k1[4], v1[4];
    if (bStart < bEnd) {
        loadKV(bStart, k0, v0);
        for (int blk = bStart; blk < bEnd; blk += 2) {
            // phase 0: refill buffer 1 (blk+1, clamped), consume buffer 0
            const int n1 = (blk + 1 < bEnd) ? blk + 1 : blk;
            loadKV(n1, k1, v1);
            __builtin_amdgcn_sched_barrier(0);
            process(blk, k0, v0);
            if (blk + 1 < bEnd) {
                // phase 1: refill buffer 0 (blk+2, clamped), consume buffer 1
                const int n2 = (blk + 2 < bEnd) ? blk + 2 : blk + 1;
                loadKV(n2, k0, v0);
                __builtin_amdgcn_sched_barrier(0);
                process(blk + 1, k1, v1);
            }
        }
    }

    // ---- flash-state merge: wave 1 publishes, wave 0 combines + stores ------
    if (wid == 1) {
#pragma unroll
        for (int nt = 0; nt < 4; ++nt)
#pragma unroll
            for (int vv = 0; vv < 8; ++vv)
                Osh[vv + half * 8][nt * 16 + lrow] = o[nt][vv];
#pragma unroll
        for (int vv = 0; vv < 8; ++vv) {       // all 16 lanes write same value
            Msh[vv + half * 8] = mrow[vv];
            Lsh[vv + half * 8] = lsum[vv];
        }
    }
    __syncthreads();
    if (wid == 0) {
#pragma unroll
        for (int vv = 0; vv < 8; ++vv) {
            const int r = vv + half * 8;
            const float mB = Msh[r], lB = Lsh[r];
            const float mnew = fmaxf(mrow[vv], mB);
            const float cA = __expf(mrow[vv] - mnew);
            const float cB = __expf(mB - mnew);
            const float linv = 1.0f / (lsum[vv] * cA + lB * cB);
#pragma unroll
            for (int nt = 0; nt < 4; ++nt)
                out[base + (size_t)(qbase + r) * HS + nt * 16 + lrow] =
                    (o[nt][vv] * cA + Osh[r][nt * 16 + lrow] * cB) * linv;
        }
    }
}

// ---------------------------------------------------------------------------
extern "C" void kernel_launch(void* const* d_in, const int* in_sizes, int n_in,
                              void* d_out, int out_size, void* d_ws, size_t ws_size,
                              hipStream_t stream) {
    const float* x  = (const float*)d_in[0];   // [B,T,E]
    const float* Wq = (const float*)d_in[1];   // [E,HS]
    const float* Wk = (const float*)d_in[2];
    const float* Wv = (const float*)d_in[3];
    float* out = (float*)d_out;                // [B,T,HS]

    char* ws = (char*)d_ws;
    __bf16* Wt = (__bf16*)ws;                                   // [3][HS][E] 384KB
    size_t off = (size_t)3 * EE * HS * 2;
    __bf16* qb = (__bf16*)(ws + off);                           // 1MB each
    __bf16* kb = (__bf16*)(ws + off + (size_t)BB * TT * HS * 2);
    __bf16* vt = (__bf16*)(ws + off + (size_t)2 * BB * TT * HS * 2);

    convw_kernel<<<(EE * HS + 255) / 256, 256, 0, stream>>>(Wq, Wk, Wv, Wt);
    // 512 row-tiles x 3 matrices = 1536 waves, 4 waves per block
    proj_kernel<<<(BB * TT / 16) * 3 / 4, 128, 0, stream>>>(x, Wt, qb, kb, vt);
    // one block (2 waves) per query tile = 1024 waves
    attn_kernel<<<BB * TT / 16, 64, 0, stream>>>(qb, kb, vt, out);
}
